// TemporalWasteGNN_84791244358296
// MI455X (gfx1250) — compile-verified, matrix operand order checked by reading, not run on hardware
//
#include <hip/hip_runtime.h>
#include <cstdint>

#define FDIM 128
#define HDIM 64
#define LHDIM 32
#define KOUT 10

typedef __bf16 v16bf __attribute__((ext_vector_type(16)));
typedef __bf16 v8bf  __attribute__((ext_vector_type(8)));
typedef float  v8f   __attribute__((ext_vector_type(8)));

// ---------------------------------------------------------------------------
// CDNA5 async global->LDS copy (CDNA5 ISA). Inline asm since
// the clang builtin arity differs across toolchains.
// ---------------------------------------------------------------------------
__device__ __forceinline__ uint32_t lds_off(const void* p) {
  // flat LDS pointers carry the LDS offset in addr[31:0] (ISA §10.2 aperture map)
  return (uint32_t)(uintptr_t)p;
}
__device__ __forceinline__ void async_b128_to_lds(uint32_t lds_byte_off, const float* gsrc) {
  asm volatile("global_load_async_to_lds_b128 %0, %1, off"
               :: "v"(lds_byte_off), "v"(gsrc) : "memory");
}
__device__ __forceinline__ void wait_async0() {
  asm volatile("s_wait_asynccnt 0" ::: "memory");
}

__device__ __forceinline__ v16bf pack16(float4 a, float4 b, float4 c, float4 d) {
  v16bf v;
  v[0]=(__bf16)a.x;  v[1]=(__bf16)a.y;  v[2]=(__bf16)a.z;  v[3]=(__bf16)a.w;
  v[4]=(__bf16)b.x;  v[5]=(__bf16)b.y;  v[6]=(__bf16)b.z;  v[7]=(__bf16)b.w;
  v[8]=(__bf16)c.x;  v[9]=(__bf16)c.y;  v[10]=(__bf16)c.z; v[11]=(__bf16)c.w;
  v[12]=(__bf16)d.x; v[13]=(__bf16)d.y; v[14]=(__bf16)d.z; v[15]=(__bf16)d.w;
  return v;
}

// ---------------------------------------------------------------------------
// C(M x NOUT) = A(M x K) @ B(K x NOUT)  (TRANSB: B element (k,n) = B[n*K+k])
// Phase 1: async-copy raw f32 B into LDS.
// Phase 2: block pre-swizzles B into bf16 WMMA fragments (ISA §7.12.2 layout).
// Phase 3: each wave computes a 16x64 output strip: A fragment loaded once per
//          K-chunk, reused across 4 WMMAs; B fragments = single 32B LDS loads.
// ABF16: A tensor stored as bf16 (direct b128 fragment loads, no cvt).
// CBF16: store C as bf16.  Requires M%16==0, K%32==0, NOUT%64==0.
// ---------------------------------------------------------------------------
template<int K, int NOUT, int TRANSB, int ABF16, int CBF16>
__global__ __launch_bounds__(256) void gemm_wmma(
    const void* __restrict__ Ap, const float* __restrict__ B,
    void* __restrict__ Cp, int M)
{
  constexpr int KC = K / 32;        // K chunks
  constexpr int NT = NOUT / 16;     // 16-wide N tiles
  constexpr int NGROUPS = NT / 4;   // 4 N-tiles per wave
  constexpr int NFRAG = NT * KC * 32;

  __shared__ float  sRaw[K * NOUT];
  __shared__ __bf16 sFrag[NFRAG * 16];

  // ---- phase 1: async global -> LDS raw copy of B -------------------------
  for (int i = threadIdx.x * 4; i < K * NOUT; i += 256 * 4)
    async_b128_to_lds(lds_off(&sRaw[i]), B + i);
  wait_async0();
  __syncthreads();

  // ---- phase 2: build bf16 B fragments in hardware layout -----------------
  for (int t = threadIdx.x; t < NFRAG; t += 256) {
    const int lane   = t & 31;
    const int fidx   = t >> 5;            // (nT*KC + kc)
    const int kc     = fidx % KC;
    const int nT     = fidx / KC;
    const int laneLo = lane & 15;
    const int laneHi = lane >> 4;
    const int n  = nT * 16 + laneLo;
    const int k0 = kc * 32 + laneHi * 8;  // elems 0..7
    const int k1 = k0 + 16;               // elems 8..15
    v16bf f;
    #pragma unroll
    for (int i = 0; i < 8; ++i)
      f[i] = (__bf16)(TRANSB ? sRaw[n * K + k0 + i] : sRaw[(k0 + i) * NOUT + n]);
    #pragma unroll
    for (int i = 0; i < 8; ++i)
      f[8 + i] = (__bf16)(TRANSB ? sRaw[n * K + k1 + i] : sRaw[(k1 + i) * NOUT + n]);
    *(v16bf*)(sFrag + (size_t)t * 16) = f;
  }
  __syncthreads();

  // ---- phase 3: WMMA ------------------------------------------------------
  const int wave   = threadIdx.x >> 5;
  const int lane   = threadIdx.x & 31;
  const int laneLo = lane & 15;
  const int laneHi = lane >> 4;
  const int mtiles = M >> 4;
  const int work   = blockIdx.x * 8 + wave;   // (mtile, ngroup)
  if (work >= mtiles * NGROUPS) return;       // wave-uniform: EXEC all-ones
  const int mt = work / NGROUPS;
  const int ng = work % NGROUPS;

  const float*  ArowF = (const float*)Ap  + (size_t)(mt * 16 + laneLo) * K;
  const __bf16* ArowB = (const __bf16*)Ap + (size_t)(mt * 16 + laneLo) * K;

  v8f acc[4];
  #pragma unroll
  for (int j = 0; j < 4; ++j) acc[j] = (v8f){0.f,0.f,0.f,0.f,0.f,0.f,0.f,0.f};

  #pragma unroll
  for (int kc = 0; kc < KC; ++kc) {
    const int k0 = kc * 32 + laneHi * 8;
    const int k1 = k0 + 16;
    v16bf a;
    if (ABF16) {
      const v8bf lo = *(const v8bf*)(ArowB + k0);     // 16B -> global_load_b128
      const v8bf hi = *(const v8bf*)(ArowB + k1);
      a = __builtin_shufflevector(lo, hi, 0,1,2,3,4,5,6,7,8,9,10,11,12,13,14,15);
    } else {
      a = pack16(*(const float4*)(ArowF + k0), *(const float4*)(ArowF + k0 + 4),
                 *(const float4*)(ArowF + k1), *(const float4*)(ArowF + k1 + 4));
    }
    #pragma unroll
    for (int j = 0; j < 4; ++j) {
      const int nT = ng * 4 + j;
      const v16bf b = *(const v16bf*)(sFrag + (size_t)((nT * KC + kc) * 32 + lane) * 16);
      acc[j] = __builtin_amdgcn_wmma_f32_16x16x32_bf16(false, a, false, b,
                                                       (short)0, acc[j], false, false);
    }
  }

  #pragma unroll
  for (int j = 0; j < 4; ++j) {
    const int n = (ng * 4 + j) * 16 + laneLo;
    #pragma unroll
    for (int r = 0; r < 8; ++r) {
      const int m = mt * 16 + r + laneHi * 8;
      if (CBF16) ((__bf16*)Cp)[(size_t)m * NOUT + n] = (__bf16)acc[j][r];
      else       ((float*)Cp)[(size_t)m * NOUT + n]  = acc[j][r];
    }
  }
}

// ---------------------------------------------------------------------------
// GCN normalization + scatter (gather side bf16, accumulation f32)
// ---------------------------------------------------------------------------
__global__ __launch_bounds__(256) void deg_kernel(const int* __restrict__ dst,
                                                  float* __restrict__ deg, int E) {
  int e = blockIdx.x * blockDim.x + threadIdx.x;
  if (e < E)
    __hip_atomic_fetch_add(&deg[dst[e]], 1.0f, __ATOMIC_RELAXED, __HIP_MEMORY_SCOPE_AGENT);
}

__global__ __launch_bounds__(256) void rsqrt_kernel(float* __restrict__ d, int N) {
  int i = blockIdx.x * blockDim.x + threadIdx.x;
  if (i < N) d[i] = rsqrtf(d[i] + 1.0f);   // +1 = self-loop; always > 0
}

__global__ __launch_bounds__(256) void scatter_kernel(
    const __bf16* __restrict__ T, const int* __restrict__ src,
    const int* __restrict__ dst, const float* __restrict__ dinv,
    float* __restrict__ agg, int E)
{
  int t = blockIdx.x * blockDim.x + threadIdx.x;
  int e = t >> 6;
  if (e >= E) return;
  int f = t & (HDIM - 1);
  int s = src[e], d = dst[e];
  float nrm = dinv[s] * dinv[d];
  float v = (float)T[(size_t)s * HDIM + f] * nrm;
  __hip_atomic_fetch_add(&agg[(size_t)d * HDIM + f], v,
                         __ATOMIC_RELAXED, __HIP_MEMORY_SCOPE_AGENT);
}

__global__ __launch_bounds__(256) void finalize_kernel(
    const float* __restrict__ agg, const __bf16* __restrict__ T,
    const float* __restrict__ dinv, const float* __restrict__ b,
    __bf16* __restrict__ H, int N)
{
  int t = blockIdx.x * blockDim.x + threadIdx.x;
  if (t >= N * HDIM) return;
  int i = t >> 6, f = t & (HDIM - 1);
  float di = dinv[i];
  H[t] = (__bf16)fmaxf(agg[t] + (float)T[t] * di * di + b[f], 0.0f);
}

// ---------------------------------------------------------------------------
// LSTM activation: c0 = 0 so only i, g, o gates matter (torch order i,f,g,o)
// ---------------------------------------------------------------------------
__device__ __forceinline__ float sigmoidf(float x) { return 1.0f / (1.0f + __expf(-x)); }

__global__ __launch_bounds__(256) void lstm_act(
    const float* __restrict__ G4, const float* __restrict__ bih,
    const float* __restrict__ bhh, __bf16* __restrict__ Hout, int N)
{
  int t = blockIdx.x * blockDim.x + threadIdx.x;
  if (t >= N * LHDIM) return;
  int node = t / LHDIM, j = t - node * LHDIM;
  const float* g = G4 + (size_t)node * 4 * LHDIM;
  float gi = g[j]             + bih[j]             + bhh[j];
  float gg = g[2*LHDIM + j]   + bih[2*LHDIM + j]   + bhh[2*LHDIM + j];
  float go = g[3*LHDIM + j]   + bih[3*LHDIM + j]   + bhh[3*LHDIM + j];
  float c = sigmoidf(gi) * tanhf(gg);
  Hout[t] = (__bf16)(sigmoidf(go) * tanhf(c));
}

// ---------------------------------------------------------------------------
// Fused prediction heads (weights staged in LDS)
// ---------------------------------------------------------------------------
__global__ __launch_bounds__(256) void heads_kernel(
    const __bf16* __restrict__ H2,
    const float* __restrict__ Wv1, const float* __restrict__ bv1,
    const float* __restrict__ Wv2, const float* __restrict__ bv2,
    const float* __restrict__ Wt1, const float* __restrict__ bt1,
    const float* __restrict__ Wt2, const float* __restrict__ bt2,
    float* __restrict__ outV, float* __restrict__ outT, int N)
{
  __shared__ float sWv1[LHDIM*LHDIM], sWt1[LHDIM*LHDIM];
  __shared__ float sWt2[LHDIM*KOUT], sWv2[LHDIM], sbv1[LHDIM], sbt1[LHDIM];
  for (int i = threadIdx.x; i < LHDIM*LHDIM; i += blockDim.x) { sWv1[i]=Wv1[i]; sWt1[i]=Wt1[i]; }
  for (int i = threadIdx.x; i < LHDIM*KOUT;  i += blockDim.x) sWt2[i]=Wt2[i];
  if (threadIdx.x < LHDIM) {
    sWv2[threadIdx.x]=Wv2[threadIdx.x]; sbv1[threadIdx.x]=bv1[threadIdx.x]; sbt1[threadIdx.x]=bt1[threadIdx.x];
  }
  __syncthreads();

  int node = blockIdx.x * blockDim.x + threadIdx.x;
  if (node >= N) return;
  float h[LHDIM];
  const __bf16* hp = H2 + (size_t)node * LHDIM;
  #pragma unroll
  for (int i = 0; i < LHDIM; ++i) h[i] = (float)hp[i];

  float vol = bv2[0];
  float ty[KOUT];
  #pragma unroll
  for (int k = 0; k < KOUT; ++k) ty[k] = bt2[k];

  for (int j = 0; j < LHDIM; ++j) {
    float av = sbv1[j], at = sbt1[j];
    #pragma unroll
    for (int i = 0; i < LHDIM; ++i) {
      av = fmaf(h[i], sWv1[i*LHDIM + j], av);
      at = fmaf(h[i], sWt1[i*LHDIM + j], at);
    }
    av = fmaxf(av, 0.f); at = fmaxf(at, 0.f);
    vol = fmaf(av, sWv2[j], vol);
    #pragma unroll
    for (int k = 0; k < KOUT; ++k) ty[k] = fmaf(at, sWt2[j*KOUT + k], ty[k]);
  }
  outV[node] = vol;
  float* tp = outT + (size_t)node * KOUT;
  #pragma unroll
  for (int k = 0; k < KOUT; ++k) tp[k] = ty[k];
}

// ---------------------------------------------------------------------------
static inline int cdiv(long a, long b) { return (int)((a + b - 1) / b); }

extern "C" void kernel_launch(void* const* d_in, const int* in_sizes, int n_in,
                              void* d_out, int out_size, void* d_ws, size_t ws_size,
                              hipStream_t stream) {
  const float* x    = (const float*)d_in[0];
  const int*   eidx = (const int*)d_in[1];
  const float* W1 = (const float*)d_in[2];  const float* b1 = (const float*)d_in[3];
  const float* W2 = (const float*)d_in[4];  const float* b2 = (const float*)d_in[5];
  const float* W3 = (const float*)d_in[6];  const float* b3 = (const float*)d_in[7];
  const float* Wih0 = (const float*)d_in[8];   // (4*LH, H) -> transB
  const float* bih0 = (const float*)d_in[10]; const float* bhh0 = (const float*)d_in[11];
  const float* Wih1 = (const float*)d_in[12];  // (4*LH, LH) -> transB
  const float* bih1 = (const float*)d_in[14]; const float* bhh1 = (const float*)d_in[15];
  const float* Wv1 = (const float*)d_in[16]; const float* bv1 = (const float*)d_in[17];
  const float* Wv2 = (const float*)d_in[18]; const float* bv2 = (const float*)d_in[19];
  const float* Wt1 = (const float*)d_in[20]; const float* bt1 = (const float*)d_in[21];
  const float* Wt2 = (const float*)d_in[22]; const float* bt2 = (const float*)d_in[23];

  const int N = in_sizes[0] / FDIM;
  const int E = in_sizes[1] / 2;
  const int* src = eidx;
  const int* dst = eidx + E;

  // workspace: [dinv f32 | T bf16 | AGG f32 | H bf16 | H2 bf16 | G f32]
  char* ws = (char*)d_ws;
  size_t o = 0;
  float*  dinv = (float*)(ws + o);  o += ((size_t)N * 4 + 255) & ~(size_t)255;
  __bf16* T    = (__bf16*)(ws + o); o += (size_t)N * HDIM * 2;
  float*  AGG  = (float*)(ws + o);  o += (size_t)N * HDIM * 4;
  __bf16* H    = (__bf16*)(ws + o); o += (size_t)N * HDIM * 2;
  __bf16* H2   = (__bf16*)(ws + o); o += (size_t)N * LHDIM * 2;
  float*  G    = (float*)(ws + o);  o += (size_t)N * 4 * LHDIM * 4;

  float* outV = (float*)d_out;
  float* outT = (float*)d_out + N;

  const int mtiles = N / 16;
  const int gemmHBlocks = cdiv((long)mtiles * 1, 8);   // NOUT=64  -> 1 ngroup
  const int gemmGBlocks = cdiv((long)mtiles * 2, 8);   // NOUT=128 -> 2 ngroups

  // normalization
  hipMemsetAsync(dinv, 0, (size_t)N * 4, stream);
  deg_kernel<<<cdiv(E, 256), 256, 0, stream>>>(dst, dinv, E);
  rsqrt_kernel<<<cdiv(N, 256), 256, 0, stream>>>(dinv, N);

  // GCN layer 1 (A = x is f32)
  gemm_wmma<FDIM, HDIM, 0, 0, 1><<<gemmHBlocks, 256, 0, stream>>>(x, W1, T, N);
  hipMemsetAsync(AGG, 0, (size_t)N * HDIM * 4, stream);
  scatter_kernel<<<cdiv((long)E * HDIM, 256), 256, 0, stream>>>(T, src, dst, dinv, AGG, E);
  finalize_kernel<<<cdiv((long)N * HDIM, 256), 256, 0, stream>>>(AGG, T, dinv, b1, H, N);

  // GCN layer 2
  gemm_wmma<HDIM, HDIM, 0, 1, 1><<<gemmHBlocks, 256, 0, stream>>>(H, W2, T, N);
  hipMemsetAsync(AGG, 0, (size_t)N * HDIM * 4, stream);
  scatter_kernel<<<cdiv((long)E * HDIM, 256), 256, 0, stream>>>(T, src, dst, dinv, AGG, E);
  finalize_kernel<<<cdiv((long)N * HDIM, 256), 256, 0, stream>>>(AGG, T, dinv, b2, H, N);

  // GCN layer 3
  gemm_wmma<HDIM, HDIM, 0, 1, 1><<<gemmHBlocks, 256, 0, stream>>>(H, W3, T, N);
  hipMemsetAsync(AGG, 0, (size_t)N * HDIM * 4, stream);
  scatter_kernel<<<cdiv((long)E * HDIM, 256), 256, 0, stream>>>(T, src, dst, dinv, AGG, E);
  finalize_kernel<<<cdiv((long)N * HDIM, 256), 256, 0, stream>>>(AGG, T, dinv, b3, H, N);

  // LSTM layer 0: G = H @ Wih0^T (f32 gates) ; activations -> H2 (bf16)
  gemm_wmma<HDIM, 4*LHDIM, 1, 1, 0><<<gemmGBlocks, 256, 0, stream>>>(H, Wih0, G, N);
  lstm_act<<<cdiv((long)N * LHDIM, 256), 256, 0, stream>>>(G, bih0, bhh0, H2, N);

  // LSTM layer 1: G = H2 @ Wih1^T ; activations -> H2 (act reads only G)
  gemm_wmma<LHDIM, 4*LHDIM, 1, 1, 0><<<gemmGBlocks, 256, 0, stream>>>(H2, Wih1, G, N);
  lstm_act<<<cdiv((long)N * LHDIM, 256), 256, 0, stream>>>(G, bih1, bhh1, H2, N);

  // heads
  heads_kernel<<<cdiv(N, 256), 256, 0, stream>>>(H2, Wv1, bv1, Wv2, bv2,
                                                 Wt1, bt1, Wt2, bt2, outV, outT, N);
}
